// LSTMFCDecoder_3358664426205
// MI455X (gfx1250) — matrix operand order repaired and checked by reading, not compiled, not run on hardware
//
#include <hip/hip_runtime.h>
#include <hip/hip_bf16.h>

typedef _Float16 f16_t;
typedef __attribute__((ext_vector_type(16))) _Float16 v16h;
typedef __attribute__((ext_vector_type(8)))  float    v8f;

struct alignas(16) H8 { f16_t h[8]; };
struct alignas(16) F4 { float x0, x1, x2, x3; };
union  V8U { v8f v; F4 q[2]; };

// ---------------- tunables / layout -----------------
#define NW 4                 // waves per block (128 threads)
// dynamic LDS layout (bytes)
#define OFF_WIH0   0         // 5 tiles * 512 halves (B-layout, K padded 20->32, bias in k=20)
#define OFF_WHH0   5120
#define OFF_WIH1   10240
#define OFF_WHH1   15360
#define OFF_B1     20480     // 80 f32 (FC1 bias)
#define OFF_W1B    20864     // 25 tiles * 512 halves (K=160 exact, 5 Kchunks x 5 Ntiles)
#define OFF_W2B    46464     // 3 tiles * 512 halves  (K padded 80->96 with bias in k=80, N padded 10->16)
#define OFF_WAVE   49536
#define WAVE_STRIDE 19456    // per-wave scratch
#define OFF_XH     0         // 8 slots * 16x32 halves (inputs + const-1 col 20; then h0[t])
#define OFF_HT     8192      // 16x32 halves (recurrent h, col 20..31 = 0)
#define OFF_G      9216      // gates col-major 80x16 f32; reused as Z1 (16x96 f16) + Z2 (16x16 f32 @ +4096)
#define OFF_HAPP   14336     // 16x160 halves (h1 concat over time, packed K)
#define SMEM_BYTES (OFF_WAVE + NW * WAVE_STRIDE)

// ---------------- helpers -----------------
__device__ __forceinline__ v8f splat8(float v) {
  v8f r;
#pragma unroll
  for (int i = 0; i < 8; ++i) r[i] = v;
  return r;
}

// A operand (16x32 f16, row-major in LDS): lane m<16 holds row m, K {kh..kh+7, 16+kh..23+kh}
__device__ __forceinline__ v16h load_a16(const f16_t* base, int rowStride, int colOff, int lane) {
  const int row = lane & 15;
  const int kh  = (lane >> 4) << 3;                 // 0 or 8
  const f16_t* p = base + row * rowStride + colOff + kh;
  union { v16h v; H8 q[2]; } u;
  u.q[0] = *(const H8*)(p);
  u.q[1] = *(const H8*)(p + 16);
  return u.v;
}

// B operand (32x16 f16, column-major tile in LDS: tile[col*32 + k])
__device__ __forceinline__ v16h load_b16(const f16_t* tile, int lane) {
  const int col = lane & 15;
  const int k0  = (lane >> 4) << 3;                 // 0 or 8
  const f16_t* p = tile + col * 32 + k0;
  union { v16h v; H8 q[2]; } u;
  u.q[0] = *(const H8*)(p);
  u.q[1] = *(const H8*)(p + 16);
  return u.v;
}

// gfx1250 has a native v_tanh_f32 TRANS op; use it for both tanh and sigmoid
__device__ __forceinline__ float tanh_fast(float v) {
#if __has_builtin(__builtin_amdgcn_tanhf)
  return __builtin_amdgcn_tanhf(v);                 // v_tanh_f32
#else
  float e = __expf(-2.0f * v);
  return 1.0f - 2.0f * __builtin_amdgcn_rcpf(1.0f + e);
#endif
}
__device__ __forceinline__ float sigf(float v) {
#if __has_builtin(__builtin_amdgcn_tanhf)
  return fmaf(0.5f, __builtin_amdgcn_tanhf(0.5f * v), 0.5f);  // 1 TRANS + 2 VALU
#else
  return __builtin_amdgcn_rcpf(1.0f + __expf(-v));
#endif
}

// One LSTM layer over 8 timesteps for a 16-row batch tile.
// Bias is folded into WiB row k=20 (input col 20 is constant 1.0).
__device__ __forceinline__ void lstm_layer(
    const f16_t* XH, f16_t* outBase, int outTS, int outRS,
    const f16_t* WiB, const f16_t* WhB,
    f16_t* htmp, float* Gf, int lane)
{
  const int col = lane & 15;
  const int hi8 = (lane >> 4) << 3;

  // keep all 10 B operands resident in VGPRs for the whole time loop
  v16h Bi[5], Bh[5];
#pragma unroll
  for (int j = 0; j < 5; ++j) {
    Bi[j] = load_b16(WiB + j * 512, lane);
    Bh[j] = load_b16(WhB + j * 512, lane);
  }

  {  // h = 0 (incl. padding cols, so recurrent matmul adds no bias)
    unsigned int* h32 = (unsigned int*)htmp;
    for (int i = lane; i < 256; i += 32) h32[i] = 0u;
  }
  float cst[10];
#pragma unroll
  for (int q = 0; q < 10; ++q) cst[q] = 0.0f;

#pragma unroll 1
  for (int t = 0; t < 8; ++t) {
    v16h Ax = load_a16(XH + t * 512, 32, 0, lane);
    v16h Ah = load_a16(htmp,         32, 0, lane);

#pragma unroll
    for (int j = 0; j < 5; ++j) {            // 5 gate chunks of 16
      v8f acc = {};                          // bias comes via k=20 row -> literal-0 C operand
      acc = __builtin_amdgcn_wmma_f32_16x16x32_f16(false, Ax, false, Bi[j], (short)0, acc, false, false);
      acc = __builtin_amdgcn_wmma_f32_16x16x32_f16(false, Ah, false, Bh[j], (short)0, acc, false, false);
      // stage gates to LDS, col-major G[n*16 + b]: lane's 8 VGPRs = 8 consecutive rows
      V8U tmp; tmp.v = acc;
      F4* gp = (F4*)(Gf + (j * 16 + col) * 16 + hi8);
      gp[0] = tmp.q[0];
      gp[1] = tmp.q[1];
    }

    // pointwise cell update: 16x20 elems, 10 per lane (same-wave LDS is in-order)
#pragma unroll
    for (int q = 0; q < 10; ++q) {
      const int flat = q * 32 + lane;
      const int b = flat / 20;
      const int k = flat % 20;
      const float gi = Gf[(k)      * 16 + b];
      const float gf = Gf[(20 + k) * 16 + b];
      const float gg = Gf[(40 + k) * 16 + b];
      const float go = Gf[(60 + k) * 16 + b];
      float c = sigf(gf) * cst[q] + sigf(gi) * tanh_fast(gg);
      cst[q] = c;
      const float h = sigf(go) * tanh_fast(c);
      const f16_t hh = (f16_t)h;
      htmp[b * 32 + k] = hh;                          // recurrent input (A layout source)
      outBase[t * outTS + b * outRS + k] = hh;        // layer output
    }
  }
}

__global__ __launch_bounds__(128) void lstm_fc_kernel(
    const float* __restrict__ y,  const float* __restrict__ x,
    const float* __restrict__ u,  const float* __restrict__ s,
    const float* __restrict__ Wih0, const float* __restrict__ Whh0,
    const float* __restrict__ bih0, const float* __restrict__ bhh0,
    const float* __restrict__ Wih1, const float* __restrict__ Whh1,
    const float* __restrict__ bih1, const float* __restrict__ bhh1,
    const float* __restrict__ W1, const float* __restrict__ b1,
    const float* __restrict__ W2, const float* __restrict__ b2,
    const float* __restrict__ W3, const float* __restrict__ b3,
    float* __restrict__ out)
{
  extern __shared__ char smem[];
  const int tid  = threadIdx.x;
  const int lane = tid & 31;
  const int wave = tid >> 5;

  f16_t* Wih0B = (f16_t*)(smem + OFF_WIH0);
  f16_t* Whh0B = (f16_t*)(smem + OFF_WHH0);
  f16_t* Wih1B = (f16_t*)(smem + OFF_WIH1);
  f16_t* Whh1B = (f16_t*)(smem + OFF_WHH1);
  float* b1s   = (float*)(smem + OFF_B1);
  f16_t* W1B   = (f16_t*)(smem + OFF_W1B);
  f16_t* W2B   = (f16_t*)(smem + OFF_W2B);

  // -------- cooperative weight reformat (once per block) --------
  {
    const float* lw[4]  = {Wih0, Whh0, Wih1, Whh1};
    f16_t*       lwd[4] = {Wih0B, Whh0B, Wih1B, Whh1B};
#pragma unroll
    for (int m = 0; m < 4; ++m) {
      const float* W = lw[m]; f16_t* D = lwd[m];
      for (int idx = tid; idx < 2560; idx += 128) {        // idx = j*512 + col*32 + k
        int j = idx >> 9, rem = idx & 511, cc = rem >> 5, k = rem & 31;
        int n = j * 16 + cc;
        float v = 0.0f;
        if (k < 20)        v = W[n * 20 + k];
        else if (k == 20) {                                // bias row (input col 20 == 1.0)
          if      (m == 0) v = bih0[n] + bhh0[n];
          else if (m == 2) v = bih1[n] + bhh1[n];
        }
        D[idx] = (f16_t)v;
      }
    }
    for (int i = tid; i < 80; i += 128) b1s[i] = b1[i];
    for (int idx = tid; idx < 12800; idx += 128) {         // (kc*5+nt)*512 + col*32 + kk
      int kc = idx / 2560, rem = idx % 2560, nt = rem >> 9;
      int r2 = rem & 511,  cc  = r2 >> 5,    kk = r2 & 31;
      W1B[idx] = (f16_t)W1[(nt * 16 + cc) * 160 + kc * 32 + kk];
    }
    for (int idx = tid; idx < 1536; idx += 128) {          // kc*512 + col*32 + kk
      int kc = idx >> 9, r = idx & 511, cc = r >> 5, kk = r & 31;
      int k = kc * 32 + kk;
      float v = 0.0f;
      if (cc < 10) {
        if (k < 80)       v = W2[cc * 80 + k];
        else if (k == 80) v = b2[cc];                      // FC2 bias row (Z1 col 80 == 1.0)
      }
      W2B[idx] = (f16_t)v;
    }
  }
  __syncthreads();

  // -------- per-wave tile --------
  char*  wbase = smem + OFF_WAVE + wave * WAVE_STRIDE;
  f16_t* XH    = (f16_t*)(wbase + OFF_XH);
  f16_t* htmp  = (f16_t*)(wbase + OFF_HT);
  float* Gf    = (float*)(wbase + OFF_G);
  f16_t* HAPP  = (f16_t*)(wbase + OFF_HAPP);
  const int tileBase = (blockIdx.x * NW + wave) * 16;

  __builtin_prefetch(y + tileBase * 8, 0, 3);
  __builtin_prefetch(x + tileBase * 80, 0, 3);
  __builtin_prefetch(u + tileBase * 40, 0, 3);
  __builtin_prefetch(s + tileBase * 32, 0, 3);

  // K-padding columns of all 8 input slots: col 20 = 1.0 (bias), cols 21..31 = 0
  {
    unsigned int* xh32 = (unsigned int*)XH;
    for (int i = lane; i < 768; i += 32) {
      int t = i / 96, rr = (i % 96) / 6, c = i % 6;
      xh32[t * 256 + rr * 16 + 10 + c] = (c == 0) ? 0x00003C00u : 0u;  // f16 1.0 in low half
    }
  }

  // gather inputs: concat [y|x|u|s] -> XH[t] tiles (f16, A-layout source rows)
#pragma unroll
  for (int q = 0; q < 10; ++q) {
    const int flat = q * 32 + lane;        // 16 rows x 20 feats
    const int b = flat / 20, k = flat % 20;
    const int gb = tileBase + b;
    const float* src; int stride;
    if (k == 0)      { src = y + gb * 8;             stride = 1;  }
    else if (k < 11) { src = x + gb * 80 + (k - 1);  stride = 10; }
    else if (k < 16) { src = u + gb * 40 + (k - 11); stride = 5;  }
    else             { src = s + gb * 32 + (k - 16); stride = 4;  }
    f16_t* dst = XH + b * 32 + k;
#pragma unroll
    for (int t = 0; t < 8; ++t) dst[t * 512] = (f16_t)src[t * stride];
  }

  // layer 0: reads XH[t], writes h0 back over XH[t] (col 20 stays 1.0 for layer-1 bias)
  lstm_layer(XH, XH, 512, 32, Wih0B, Whh0B, htmp, Gf, lane);
  // layer 1: reads h0 from XH[t], writes h1 packed into HAPP (k = t*20 + h)
  lstm_layer(XH, HAPP, 20, 160, Wih1B, Whh1B, htmp, Gf, lane);

  // -------- FC head --------
  const int col = lane & 15;
  const int hi8 = (lane >> 4) << 3;
  f16_t* Z1h = (f16_t*)Gf;                       // 16 x 96 halves (col 80 = 1.0, 81..95 = 0)
  for (int i = lane; i < 128; i += 32) {
    int rr = i >> 3, c = i & 7;
    *(unsigned int*)((char*)Z1h + rr * 192 + 160 + c * 4) = (c == 0) ? 0x00003C00u : 0u;
  }

  // FC1: [16,160] x [160,80], 5 Kchunks x 5 Ntiles (bias via splat C)
  v16h Afc[5];
#pragma unroll
  for (int kc = 0; kc < 5; ++kc) Afc[kc] = load_a16(HAPP, 160, kc * 32, lane);
#pragma unroll
  for (int nt = 0; nt < 5; ++nt) {
    v8f acc = splat8(b1s[nt * 16 + col]);
#pragma unroll
    for (int kc = 0; kc < 5; ++kc)
      acc = __builtin_amdgcn_wmma_f32_16x16x32_f16(false, Afc[kc], false,
              load_b16(W1B + (kc * 5 + nt) * 512, lane), (short)0, acc, false, false);
#pragma unroll
    for (int v = 0; v < 8; ++v)
      Z1h[(v + hi8) * 96 + nt * 16 + col] = (f16_t)fmaxf(acc[v], 0.0f);   // ReLU
  }

  // FC2: [16,96pad] x [96,16pad], bias via k=80 row -> zero C
  float* Z2f = (float*)((char*)Gf + 4096);       // col-major 16x16 f32: Z2f[n*16 + b]
  {
    v16h Az[3];
#pragma unroll
    for (int kc = 0; kc < 3; ++kc) Az[kc] = load_a16(Z1h, 96, kc * 32, lane);
    v8f acc = {};
#pragma unroll
    for (int kc = 0; kc < 3; ++kc)
      acc = __builtin_amdgcn_wmma_f32_16x16x32_f16(false, Az[kc], false,
              load_b16(W2B + kc * 512, lane), (short)0, acc, false, false);
    v8f r;
#pragma unroll
    for (int v = 0; v < 8; ++v) r[v] = fmaxf(acc[v], 0.0f);               // ReLU
    V8U tmp; tmp.v = r;
    F4* zp = (F4*)(Z2f + col * 16 + hi8);
    zp[0] = tmp.q[0];
    zp[1] = tmp.q[1];
  }

  // FC3: tiny dot (10) per batch row — VALU, lanes 0..15
  if (lane < 16) {
    float sum = b3[0];
#pragma unroll
    for (int k = 0; k < 10; ++k) sum += Z2f[k * 16 + lane] * W3[k];
    out[tileBase + lane] = sum;
  }
}

extern "C" void kernel_launch(void* const* d_in, const int* in_sizes, int n_in,
                              void* d_out, int out_size, void* d_ws, size_t ws_size,
                              hipStream_t stream) {
  const float* y    = (const float*)d_in[0];
  const float* x    = (const float*)d_in[1];
  const float* u    = (const float*)d_in[2];
  const float* s    = (const float*)d_in[3];
  const float* Wih0 = (const float*)d_in[4];
  const float* Whh0 = (const float*)d_in[5];
  const float* bih0 = (const float*)d_in[6];
  const float* bhh0 = (const float*)d_in[7];
  const float* Wih1 = (const float*)d_in[8];
  const float* Whh1 = (const float*)d_in[9];
  const float* bih1 = (const float*)d_in[10];
  const float* bhh1 = (const float*)d_in[11];
  const float* W1   = (const float*)d_in[12];
  const float* b1   = (const float*)d_in[13];
  const float* W2   = (const float*)d_in[14];
  const float* b2   = (const float*)d_in[15];
  const float* W3   = (const float*)d_in[16];
  const float* b3   = (const float*)d_in[17];

  const int B = in_sizes[0] / 8;          // y_past is [B, 8, 1]
  const int nTiles = B / 16;
  const int blocks = nTiles / NW;         // 4096 for B = 262144

  lstm_fc_kernel<<<dim3(blocks), dim3(128), SMEM_BYTES, stream>>>(
      y, x, u, s, Wih0, Whh0, bih0, bhh0, Wih1, Whh1, bih1, bhh1,
      W1, b1, W2, b2, W3, b3, (float*)d_out);
}
